// HeteroGraphEncoder_44040594653888
// MI455X (gfx1250) — matrix operand order repaired and checked by reading, not compiled, not run on hardware
//
#include <hip/hip_runtime.h>
#include <hip/hip_bf16.h>
#include <math.h>

typedef __attribute__((ext_vector_type(16))) _Float16 v16h;
typedef __attribute__((ext_vector_type(8)))  _Float16 v8h;
typedef __attribute__((ext_vector_type(8)))  float    v8f;

#define BB    2048
#define DD    256
#define HH    8
#define LL    4
#define DHD   32
#define NODES 12
#define NTOT  (BB*NODES)

union FragH { v16h v; v8h h[2]; };

__device__ __forceinline__ float gelu_exact(float x) {
    return 0.5f * x * (1.0f + erff(x * 0.70710678118654752f));
}

// blockDim.x must be 256
__device__ __forceinline__ float block_reduce_sum(float v, float* red) {
    int t = threadIdx.x;
    red[t] = v; __syncthreads();
    for (int s = 128; s > 0; s >>= 1) {
        if (t < s) red[t] += red[t + s];
        __syncthreads();
    }
    float r = red[0]; __syncthreads();
    return r;
}

// ---------------------------------------------------------------- concat x
__global__ void concat_kernel(const float* __restrict__ patch,
                              const float* __restrict__ band,
                              const float* __restrict__ summ,
                              float* __restrict__ x) {
    int node = blockIdx.x, t = threadIdx.x;
    int b = node / NODES, n = node % NODES;
    float v;
    if (n < 6)       v = patch[(size_t)(b * 6 + n) * DD + t];
    else if (n < 11) v = band[(size_t)(b * 5 + (n - 6)) * DD + t];
    else             v = summ[t];
    x[(size_t)node * DD + t] = v;
}

// ------------------------------------------- weight f32 -> f16, transposed
// W row-major K x Nn ; Wt[n*K + k] = (f16)W[k*Nn + n]
__global__ void w_transpose_f16(const float* __restrict__ W,
                                _Float16* __restrict__ Wt, int K, int Nn) {
    int idx = blockIdx.x * blockDim.x + threadIdx.x;
    if (idx >= K * Nn) return;
    int n = idx / K, k = idx % K;
    Wt[idx] = (_Float16)W[(size_t)k * Nn + n];
}

// ---------------------------------------------------------- LayerNorm->f16
__global__ void ln_to_f16(const float* __restrict__ x,
                          const float* __restrict__ s,
                          const float* __restrict__ bsh,
                          _Float16* __restrict__ h) {
    __shared__ float red[256];
    int row = blockIdx.x, t = threadIdx.x;
    float v = x[(size_t)row * DD + t];
    float mu  = block_reduce_sum(v, red) * (1.0f / DD);
    float d   = v - mu;
    float var = block_reduce_sum(d * d, red) * (1.0f / DD);
    float rstd = rsqrtf(var + 1e-5f);
    h[(size_t)row * DD + t] = (_Float16)(d * rstd * s[t] + bsh[t]);
}

// ----------------------------------------------------------- WMMA GEMM
// C[M x Nn] = act( A[M x K] @ W[K x Nn] + bias (+ resid) )
// A: f16 row-major. Wt: f16 transposed layout Wt[n*K + k].
// Block = 256 threads = 8 waves. Block tile = 32 rows x 256 cols.
// Wave w: row-tile rt = w>>2 (16 rows), col-strip ct = w&3 (64 cols = 4 WMMA
// tiles, 4 accumulators reusing one A fragment per K-step).
// K loop manually unrolled 2x with ping-pong fragment sets (no rotation
// copies). All four B tiles are addressed from ONE base pointer with
// constant element offsets j*16*K (fits the 24-bit signed IOFFSET), so no
// extra 64-bit pointers stay live across the loop (avoids scratch spills).
// grid.x = M/32, grid.y = Nn/256. Dynamic LDS = 32*K*2 bytes (A tile),
// staged via gfx1250 GLOBAL_LOAD_ASYNC_TO_LDS_B128 (ASYNCcnt).
__global__ void gemm_f16_wmma(const _Float16* __restrict__ A,
                              const _Float16* __restrict__ Wt,
                              const float* __restrict__ bias,
                              const float* __restrict__ resid,
                              float* __restrict__ Cf,
                              _Float16* __restrict__ Ch,
                              int K, int Nn, int act) {
    extern __shared__ _Float16 lA[];
    int tid = threadIdx.x;
    int mt  = blockIdx.x;

    // ---- async-stage contiguous 32 x K A tile into LDS (16B granules)
    {
        const _Float16* abase = A + (size_t)mt * 32 * K;   // uniform -> SGPR pair
        unsigned lbase = (unsigned)(size_t)(void*)lA;      // LDS offset (low 32b)
        int tot16 = (32 * K) >> 3;                         // # of 16-byte chunks
        for (int i = tid; i < tot16; i += 256) {
            unsigned ldsoff = lbase + (unsigned)(i * 16);
            unsigned goff   = (unsigned)(i * 16);
            asm volatile("global_load_async_to_lds_b128 %0, %1, %2"
                         :: "v"(ldsoff), "v"(goff), "s"(abase) : "memory");
        }
        asm volatile("s_wait_asynccnt 0x0" ::: "memory");
    }
    __syncthreads();

    int lane = tid & 31, wv = tid >> 5;
    int rt = wv >> 2;                          // row tile within block (0..1)
    int ct = wv & 3;                           // 64-col strip within block
    int nl = lane & 15;                        // A: M row / B,C: N col
    int kh = lane >> 4;                        // K half select
    int nbBase = blockIdx.y * 256 + ct * 64;   // first output column of strip

    v8f c0, c1, c2, c3;
    {
        float b0 = bias ? bias[nbBase + 0  + nl] : 0.0f;
        float b1 = bias ? bias[nbBase + 16 + nl] : 0.0f;
        float b2 = bias ? bias[nbBase + 32 + nl] : 0.0f;
        float b3 = bias ? bias[nbBase + 48 + nl] : 0.0f;
        for (int r = 0; r < 8; ++r) { c0[r] = b0; c1[r] = b1; c2[r] = b2; c3[r] = b3; }
    }

    // Single B base pointer; tile j lives at constant element offset j*16*K.
    const _Float16* wbase = Wt + (size_t)(nbBase + nl) * K;
    const size_t js = (size_t)16 * K;          // element stride between B tiles
    const _Float16* arow = lA + (size_t)(rt * 16 + nl) * K;

    // Fragment loaders.
    // A: lane m=nl; elems 0..7 -> K=k0+kh*8+j, elems 8..15 -> K=k0+16+kh*8+j.
    // B: lane n=nl; elems 0..7 -> K=k0+kh*8+j, elems 8..15 -> K=k0+16+kh*8+j.
    #define LOAD_A(f, k0)                                            \
        (f).h[0] = *(const v8h*)(arow + (k0) + kh * 8);              \
        (f).h[1] = *(const v8h*)(arow + (k0) + 16 + kh * 8);
    #define LOAD_B(f, off)                                           \
        { const v8h* _p = (const v8h*)(wbase + (off) + kh * 8);      \
          (f).h[0] = _p[0]; (f).h[1] = _p[2]; }
    #define WMMA4(af, f0, f1, f2, f3)                                                                        \
        c0 = __builtin_amdgcn_wmma_f32_16x16x32_f16(false, (af).v, false, (f0).v, (short)0, c0, false, false); \
        c1 = __builtin_amdgcn_wmma_f32_16x16x32_f16(false, (af).v, false, (f1).v, (short)0, c1, false, false); \
        c2 = __builtin_amdgcn_wmma_f32_16x16x32_f16(false, (af).v, false, (f2).v, (short)0, c2, false, false); \
        c3 = __builtin_amdgcn_wmma_f32_16x16x32_f16(false, (af).v, false, (f3).v, (short)0, c3, false, false);

    FragH a0, p0, p1, p2, p3;      // ping set
    FragH a1, q0, q1, q2, q3;      // pong set
    LOAD_A(a0, 0);
    LOAD_B(p0, 0);      LOAD_B(p1, js);
    LOAD_B(p2, 2 * js); LOAD_B(p3, 3 * js);

    for (int k0 = 0; k0 < K; k0 += 64) {
        size_t k1 = (size_t)k0 + 32;
        // load pong set while ping set computes
        LOAD_A(a1, k1);
        LOAD_B(q0, k1);          LOAD_B(q1, k1 + js);
        LOAD_B(q2, k1 + 2 * js); LOAD_B(q3, k1 + 3 * js);
        if (k0 + 128 < K) __builtin_prefetch(wbase + k0 + 128, 0, 1);
        WMMA4(a0, p0, p1, p2, p3)
        size_t k2 = (size_t)k0 + 64;
        if (k2 < (size_t)K) {
            // reload ping set while pong set computes
            LOAD_A(a0, k2);
            LOAD_B(p0, k2);          LOAD_B(p1, k2 + js);
            LOAD_B(p2, k2 + 2 * js); LOAD_B(p3, k2 + 3 * js);
        }
        WMMA4(a1, q0, q1, q2, q3)
    }
    #undef LOAD_A
    #undef LOAD_B
    #undef WMMA4

    // C layout: lane 0-15 -> N=lane, M=r ; lane 16-31 -> N=lane-16, M=8+r
    size_t rowbase = (size_t)mt * 32 + rt * 16;
    int mo = kh * 8;
    for (int j = 0; j < 4; ++j) {
        v8f* cj = (j == 0) ? &c0 : (j == 1) ? &c1 : (j == 2) ? &c2 : &c3;
        int nb = nbBase + j * 16;
        for (int r = 0; r < 8; ++r) {
            size_t off = (rowbase + mo + r) * (size_t)Nn + nb + nl;
            float val = (*cj)[r];
            if (resid) val += resid[off];
            if (act)   val = gelu_exact(val);
            if (Cf) Cf[off] = val;
            if (Ch) Ch[off] = (_Float16)val;
        }
    }
}

// --------------------------------------------------- dense 12x12 attention
__global__ void attn_kernel(const float* __restrict__ q,
                            const float* __restrict__ k,
                            const float* __restrict__ v,
                            _Float16* __restrict__ a16,
                            unsigned allow) {
    __shared__ float sq[NODES * DD], sk[NODES * DD], sv[NODES * DD];
    int bidx = blockIdx.x, tid = threadIdx.x;
    size_t base = (size_t)bidx * NODES * DD;
    for (int i = tid; i < NODES * DD; i += blockDim.x) {
        sq[i] = q[base + i]; sk[i] = k[base + i]; sv[i] = v[base + i];
    }
    __syncthreads();
    if (tid < NODES * HH) {
        int dst = tid >> 3, h = tid & 7;
        const float scale = 0.17677669529663687f;   // 1/sqrt(32)
        int cd = dst < 6 ? 0 : (dst < 11 ? 1 : 2);
        float sc[NODES];
        float mx = -1e30f;
        const float* qp = sq + dst * DD + h * DHD;
        for (int src = 0; src < NODES; ++src) {
            int cs = src < 6 ? 0 : (src < 11 ? 1 : 2);
            int et = cs * 3 + cd;
            float eb = ((allow >> et) & 1u) ? 0.0f : -1e9f;
            const float* kp = sk + src * DD + h * DHD;
            float d = 0.f;
            for (int i = 0; i < DHD; ++i) d += qp[i] * kp[i];
            sc[src] = d * scale + eb;
            mx = fmaxf(mx, sc[src]);
        }
        float den = 0.f, acc[DHD];
        for (int i = 0; i < DHD; ++i) acc[i] = 0.f;
        for (int src = 0; src < NODES; ++src) {
            float e = __expf(sc[src] - mx);
            den += e;
            const float* vp = sv + src * DD + h * DHD;
            for (int i = 0; i < DHD; ++i) acc[i] += e * vp[i];
        }
        float inv = 1.0f / (den + 1e-9f);
        _Float16* op = a16 + base + dst * DD + h * DHD;
        for (int i = 0; i < DHD; ++i) op[i] = (_Float16)(acc[i] * inv);
    }
}

// --------------------------------------- gated pooling + proj + LN + GELU
__global__ void final_kernel(const float* __restrict__ x,
                             const float* __restrict__ gw_p, const float* __restrict__ gb_p,
                             const float* __restrict__ gw_b, const float* __restrict__ gb_b,
                             const float* __restrict__ projw, const float* __restrict__ projb,
                             const float* __restrict__ plns, const float* __restrict__ plnb,
                             float* __restrict__ out) {
    __shared__ float red[256];
    __shared__ float cbuf[768];
    __shared__ float gates[11];
    int b = blockIdx.x, t = threadIdx.x;
    size_t base = (size_t)b * NODES * DD;
    for (int n = 0; n < 11; ++n) {
        const float* gw = (n < 6) ? gw_p : gw_b;
        float part = x[base + n * DD + t] * gw[t];
        float s = block_reduce_sum(part, red);
        if (t == 0) {
            float gb = (n < 6) ? gb_p[0] : gb_b[0];
            gates[n] = 1.0f / (1.0f + __expf(-(s + gb)));
        }
        __syncthreads();
    }
    cbuf[t] = x[base + 11 * DD + t];
    float pp = 0.f; for (int n = 0; n < 6;  ++n) pp += gates[n] * x[base + n * DD + t];
    float bp = 0.f; for (int n = 6; n < 11; ++n) bp += gates[n] * x[base + n * DD + t];
    cbuf[256 + t] = pp;
    cbuf[512 + t] = bp;
    __syncthreads();
    float acc = projb[t];
    for (int j = 0; j < 768; ++j) acc += cbuf[j] * projw[(size_t)j * 256 + t];
    float mu  = block_reduce_sum(acc, red) * (1.0f / 256.0f);
    float d   = acc - mu;
    float var = block_reduce_sum(d * d, red) * (1.0f / 256.0f);
    float y = d * rsqrtf(var + 1e-5f) * plns[t] + plnb[t];
    out[(size_t)b * DD + t] = gelu_exact(y);
}

// ------------------------------------------------------------------- host
extern "C" void kernel_launch(void* const* d_in, const int* in_sizes, int n_in,
                              void* d_out, int out_size, void* d_ws, size_t ws_size,
                              hipStream_t stream) {
    (void)in_sizes; (void)n_in; (void)out_size; (void)ws_size;
    const float* patch = (const float*)d_in[0];
    const float* band  = (const float*)d_in[1];
    const float* summ  = (const float*)d_in[2];
    const float* ln1s  = (const float*)d_in[3];
    const float* ln1b  = (const float*)d_in[4];
    const float* Wq    = (const float*)d_in[5];
    const float* bq    = (const float*)d_in[6];
    const float* Wk    = (const float*)d_in[7];
    const float* bk    = (const float*)d_in[8];
    const float* Wv    = (const float*)d_in[9];
    const float* bv    = (const float*)d_in[10];
    const float* Wo    = (const float*)d_in[11];
    const float* bo    = (const float*)d_in[12];
    const float* ln2s  = (const float*)d_in[13];
    const float* ln2b  = (const float*)d_in[14];
    const float* W1    = (const float*)d_in[15];
    const float* b1    = (const float*)d_in[16];
    const float* W2    = (const float*)d_in[17];
    const float* b2    = (const float*)d_in[18];
    const float* gw_p  = (const float*)d_in[19];
    const float* gb_p  = (const float*)d_in[20];
    const float* gw_b  = (const float*)d_in[21];
    const float* gb_b  = (const float*)d_in[22];
    const float* projw = (const float*)d_in[23];
    const float* projb = (const float*)d_in[24];
    const float* plns  = (const float*)d_in[25];
    const float* plnb  = (const float*)d_in[26];
    float* out = (float*)d_out;

    // ---- workspace carve (256B aligned)
    char* ws = (char*)d_ws;
    auto carve = [&](size_t bytes) -> void* {
        void* p = (void*)ws;
        ws += (bytes + 255) & ~(size_t)255;
        return p;
    };
    float*    x     = (float*)   carve((size_t)NTOT * DD * 4);
    _Float16* h16   = (_Float16*)carve((size_t)NTOT * DD * 2);
    float*    qb    = (float*)   carve((size_t)NTOT * DD * 4);
    float*    kb    = (float*)   carve((size_t)NTOT * DD * 4);
    float*    vb    = (float*)   carve((size_t)NTOT * DD * 4);
    _Float16* a16   = (_Float16*)carve((size_t)NTOT * DD * 2);
    _Float16* mid16 = (_Float16*)carve((size_t)NTOT * DD * 4 * 2);
    _Float16* wq16  = (_Float16*)carve((size_t)LL * DD * DD * 2);
    _Float16* wk16  = (_Float16*)carve((size_t)LL * DD * DD * 2);
    _Float16* wv16  = (_Float16*)carve((size_t)LL * DD * DD * 2);
    _Float16* wo16  = (_Float16*)carve((size_t)LL * DD * DD * 2);
    _Float16* w116  = (_Float16*)carve((size_t)LL * DD * 1024 * 2);
    _Float16* w216  = (_Float16*)carve((size_t)LL * 1024 * DD * 2);

    // ---- build x
    concat_kernel<<<NTOT, 256, 0, stream>>>(patch, band, summ, x);

    // ---- convert + transpose all weights to f16
    const int eDD  = DD * DD;       // 65536
    const int eFF  = DD * 1024;     // 262144
    for (int l = 0; l < LL; ++l) {
        w_transpose_f16<<<(eDD + 255) / 256, 256, 0, stream>>>(Wq + (size_t)l * eDD, wq16 + (size_t)l * eDD, DD, DD);
        w_transpose_f16<<<(eDD + 255) / 256, 256, 0, stream>>>(Wk + (size_t)l * eDD, wk16 + (size_t)l * eDD, DD, DD);
        w_transpose_f16<<<(eDD + 255) / 256, 256, 0, stream>>>(Wv + (size_t)l * eDD, wv16 + (size_t)l * eDD, DD, DD);
        w_transpose_f16<<<(eDD + 255) / 256, 256, 0, stream>>>(Wo + (size_t)l * eDD, wo16 + (size_t)l * eDD, DD, DD);
        w_transpose_f16<<<(eFF + 255) / 256, 256, 0, stream>>>(W1 + (size_t)l * eFF, w116 + (size_t)l * eFF, DD, 1024);
        w_transpose_f16<<<(eFF + 255) / 256, 256, 0, stream>>>(W2 + (size_t)l * eFF, w216 + (size_t)l * eFF, 1024, DD);
    }

    // pathway masks per layer: bit e set => edge-type e allowed (bias 0)
    static const unsigned masks[LL] = {0x1FFu, 0x111u, 0x1E4u, 0x1FFu};

    dim3 gQ(NTOT / 32, 1);   // Nn = 256
    dim3 gF1(NTOT / 32, 4);  // Nn = 1024, K = 256
    const size_t smQ  = 32 * 256 * 2;    // A tile K=256  (16 KB)
    const size_t smF2 = 32 * 1024 * 2;   // A tile K=1024 (64 KB)

    for (int l = 0; l < LL; ++l) {
        ln_to_f16<<<NTOT, 256, 0, stream>>>(x, ln1s + l * DD, ln1b + l * DD, h16);
        gemm_f16_wmma<<<gQ, 256, smQ, stream>>>(h16, wq16 + (size_t)l * eDD, bq + l * DD,
                                                nullptr, qb, nullptr, 256, 256, 0);
        gemm_f16_wmma<<<gQ, 256, smQ, stream>>>(h16, wk16 + (size_t)l * eDD, bk + l * DD,
                                                nullptr, kb, nullptr, 256, 256, 0);
        gemm_f16_wmma<<<gQ, 256, smQ, stream>>>(h16, wv16 + (size_t)l * eDD, bv + l * DD,
                                                nullptr, vb, nullptr, 256, 256, 0);
        attn_kernel<<<BB, 128, 0, stream>>>(qb, kb, vb, a16, masks[l]);
        gemm_f16_wmma<<<gQ, 256, smQ, stream>>>(a16, wo16 + (size_t)l * eDD, bo + l * DD,
                                                x, x, nullptr, 256, 256, 0);
        ln_to_f16<<<NTOT, 256, 0, stream>>>(x, ln2s + l * DD, ln2b + l * DD, h16);
        gemm_f16_wmma<<<gF1, 256, smQ, stream>>>(h16, w116 + (size_t)l * eFF, b1 + l * 1024,
                                                 nullptr, nullptr, mid16, 256, 1024, 1);
        gemm_f16_wmma<<<gQ, 256, smF2, stream>>>(mid16, w216 + (size_t)l * eFF, b2 + l * DD,
                                                 x, x, nullptr, 1024, 256, 0);
    }

    final_kernel<<<BB, 256, 0, stream>>>(x, gw_p, gb_p, gw_b, gb_b,
                                         projw, projb, plns, plnb, out);
}
